// GATKhopEncoder_88149908783554
// MI455X (gfx1250) — matrix operand order repaired and checked by reading, not compiled, not run on hardware
//
#include <hip/hip_runtime.h>
#include <hip/hip_bf16.h>
#include <cstddef>

// ---------------------------------------------------------------------------
// GAT k-hop encoder for MI455X (gfx1250, wave32, WMMA).
// N=4096, CIN=128, CH=64, COUT=32, T=3.
// Bandwidth-bound on M_denses/adj (~0.87 GB min traffic -> ~37us @ 23.3TB/s).
// Each (hop,layer) streams M/adj exactly once in a fused
// score->softmax->P@H pass; all matmuls use v_wmma_f32_16x16x32_f16.
// This revision: replace IEEE division chains (v_div_scale/v_div_fmas) with
// single-instruction v_rcp_f32 via __builtin_amdgcn_rcpf in the softmax and
// sigmoid epilogues (16.7M sigmoids); rcp is a TRANS op that co-executes
// with VALU on CDNA5.
// ---------------------------------------------------------------------------

typedef _Float16 v16h __attribute__((ext_vector_type(16)));
typedef _Float16 v8h  __attribute__((ext_vector_type(8)));
typedef _Float16 v4h  __attribute__((ext_vector_type(4)));
typedef float    v8f  __attribute__((ext_vector_type(8)));

#define NROWS 4096
#define ALPHA 0.2f
#define EPSN  1e-12f

__device__ __forceinline__ v8f wmma_f16(const v16h& a, const v16h& b, const v8f& c) {
    return __builtin_amdgcn_wmma_f32_16x16x32_f16(false, a, false, b, (short)0, c, false, false);
}

// single v_rcp_f32 (TRANS), no Newton refinement chain
__device__ __forceinline__ float fastrcp(float x) { return __builtin_amdgcn_rcpf(x); }

// exp-score: leaky((s_i + n_j) * M_ij), adj-masked (mask == exp(-9e15) -> 0)
__device__ __forceinline__ float pexp(float sv, float nv, float m, float a) {
    float val = (sv + nv) * m;
    val = val > 0.f ? val : ALPHA * val;
    return (a > 0.f) ? __expf(val) : 0.f;
}

// ---------------------------------------------------------------------------
// Dense GEMM H = A[f32, N x KD] @ W[f32, KD x ND], f32 accum via f16 WMMA.
// One wave per 16x16 tile; block of ND/16 waves covers a 16-row stripe.
// A-row loads are contiguous -> compiler clauses them into global_load_b128.
// ---------------------------------------------------------------------------
template<int KD, int ND>
__global__ void gemm_wmma(const float* __restrict__ Ain, const float* __restrict__ W,
                          float* __restrict__ Hf32, _Float16* __restrict__ Hf16)
{
    const int lane = threadIdx.x & 31;
    const int wave = threadIdx.x >> 5;
    const int i0   = blockIdx.x * 16;
    const int m    = lane & 15;      // A row / B col within tile
    const int half = lane >> 4;      // lane group 0/1

    v8f acc = {};
    for (int k0 = 0; k0 < KD; k0 += 32) {
        // A operand: row m, contiguous K = k0 + half*16 .. +15
        const float* ar = Ain + (size_t)(i0 + m) * KD + k0 + half * 16;
        v16h a;
        #pragma unroll
        for (int t = 0; t < 16; ++t) a[t] = (_Float16)ar[t];
        // B operand: column (wave*16 + m), contiguous K = k0 + half*16 .. +15
        const int c  = wave * 16 + m;
        const int kb = k0 + half * 16;
        v16h b;
        #pragma unroll
        for (int t = 0; t < 16; ++t) b[t] = (_Float16)W[(size_t)(kb + t) * ND + c];
        acc = wmma_f16(a, b, acc);
    }
    #pragma unroll
    for (int r = 0; r < 8; ++r) {
        const int orow = i0 + r + (half ? 8 : 0);
        const int ocol = wave * 16 + m;
        const float v  = acc[r];
        Hf32[(size_t)orow * ND + ocol] = v;
        Hf16[(size_t)orow * ND + ocol] = (_Float16)v;
    }
}

// ---------------------------------------------------------------------------
// Per-row score dots: s[i] = h[i,:]·a_s, n[i] = h[i,:]·a_n  (tiny)
// ---------------------------------------------------------------------------
template<int D>
__global__ void row_dots(const float* __restrict__ H, const float* __restrict__ a_s,
                         const float* __restrict__ a_n,
                         float* __restrict__ s, float* __restrict__ n)
{
    const int i = blockIdx.x * blockDim.x + threadIdx.x;
    if (i >= NROWS) return;
    float ss = 0.f, nn = 0.f;
    #pragma unroll 8
    for (int c = 0; c < D; ++c) {
        const float h = H[(size_t)i * D + c];
        ss += h * a_s[c];
        nn += h * a_n[c];
    }
    s[i] = ss; n[i] = nn;
}

// ---------------------------------------------------------------------------
// Fused attention layer: one streaming pass over M/adj per 16-row stripe.
// Software-pipelined: M/adj tile (j+1) is loaded into registers (b128) while
// tile j runs exp + LDS staging + WMMA. Softmax denominators accumulate via
// one ds_add_f32 per 8-lane row group. Epilogue: softmax scale + elu, then
// write G (mode 0) or init/accumulate the hop sum (mode 1/2).
// ---------------------------------------------------------------------------
template<int DV>
__global__ void attn_wmma(const float* __restrict__ Mh, const float* __restrict__ adj,
                          const float* __restrict__ svec, const float* __restrict__ nvec,
                          const _Float16* __restrict__ Hf16,
                          float* __restrict__ outF32, _Float16* __restrict__ outF16,
                          float* __restrict__ accum, int mode)
{
    constexpr int NTHR   = 32 * (DV / 16);
    constexpr int NCHUNK = 512 / (NTHR * 4);          // float4 chunks per thread
    constexpr int HCHUNK = (DV * 32 / 8) / NTHR;      // v8h chunks per thread

    __shared__ __align__(32) _Float16 sP[16 * 32];
    __shared__ __align__(32) _Float16 sHT[DV * 32];
    __shared__ float sS[16];
    __shared__ float sSum[16];

    const int tid  = threadIdx.x;
    const int lane = tid & 31;
    const int wave = tid >> 5;
    const int i0   = blockIdx.x * 16;
    const int m    = lane & 15;
    const int half = lane >> 4;

    if (tid < 16) { sS[tid] = svec[i0 + tid]; sSum[tid] = 0.f; }
    __syncthreads();

    float4 mreg[NCHUNK], areg[NCHUNK];
    auto load_tiles = [&](int j0n) {
        #pragma unroll
        for (int q = 0; q < NCHUNK; ++q) {
            const int e0 = (tid + q * NTHR) * 4;
            const int r  = e0 >> 5, c = e0 & 31;
            const size_t base = (size_t)(i0 + r) * NROWS + j0n + c;
            mreg[q] = *(const float4*)&Mh[base];      // global_load_b128
            areg[q] = *(const float4*)&adj[base];
        }
    };
    load_tiles(0);

    v8f acc = {};
    for (int j0 = 0; j0 < NROWS; j0 += 32) {
        // ---- P tile from prefetched registers ----
        #pragma unroll
        for (int q = 0; q < NCHUNK; ++q) {
            const int e0 = (tid + q * NTHR) * 4;
            const int r  = e0 >> 5, c = e0 & 31;
            const float4 m4 = mreg[q], a4 = areg[q];
            const float4 nv4 = *(const float4*)&nvec[j0 + c];
            const float sv = sS[r];
            const float p0 = pexp(sv, nv4.x, m4.x, a4.x);
            const float p1 = pexp(sv, nv4.y, m4.y, a4.y);
            const float p2 = pexp(sv, nv4.z, m4.z, a4.z);
            const float p3 = pexp(sv, nv4.w, m4.w, a4.w);
            const v4h ph = { (_Float16)p0, (_Float16)p1, (_Float16)p2, (_Float16)p3 };
            *(v4h*)&sP[r * 32 + c] = ph;
            // 8-lane group shares one softmax row -> reduce, 1 atomic/group
            float part = (p0 + p1) + (p2 + p3);
            part += __shfl_down(part, 4, 8);
            part += __shfl_down(part, 2, 8);
            part += __shfl_down(part, 1, 8);
            if ((tid & 7) == 0) atomicAdd(&sSum[r], part);   // ds_add_f32
        }
        // ---- kick off next M/adj tile loads (overlap with LDS+WMMA) ----
        if (j0 + 32 < NROWS) load_tiles(j0 + 32);
        // ---- H value tile, transposed into B-operand layout ----
        #pragma unroll
        for (int q = 0; q < HCHUNK; ++q) {
            const int idx = tid + q * NTHR;
            const int jj  = idx / (DV / 8);
            const int c0  = (idx % (DV / 8)) * 8;
            const v8h hv = *(const v8h*)&Hf16[(size_t)(j0 + jj) * DV + c0];
            #pragma unroll
            for (int t = 0; t < 8; ++t) sHT[(c0 + t) * 32 + jj] = hv[t];
        }
        __syncthreads();
        // ---- WMMA: acc += P(16x32) @ H(32x16 per wave) ----
        v16h a;
        const _Float16* pr = &sP[m * 32 + half * 16];
        #pragma unroll
        for (int t = 0; t < 16; ++t) a[t] = pr[t];
        const v16h b = *(const v16h*)&sHT[(wave * 16 + m) * 32 + half * 16];
        acc = wmma_f16(a, b, acc);
        __syncthreads();   // protect sP/sHT reuse
    }

    // ---- epilogue: softmax normalize (v_rcp_f32), elu, write/accumulate ----
    const int rbase = half ? 8 : 0;
    const int ocol  = wave * 16 + m;
    #pragma unroll
    for (int r = 0; r < 8; ++r) {
        const int row = r + rbase;
        float denom = sSum[row];
        denom = denom > 1e-30f ? denom : 1e-30f;
        const float att = acc[r] * fastrcp(denom);
        const float g   = att > 0.f ? att : (__expf(att) - 1.f);   // elu
        const size_t idx = (size_t)(i0 + row) * DV + ocol;
        if (mode == 0)      { outF32[idx] = g; outF16[idx] = (_Float16)g; }
        else if (mode == 1) { accum[idx]  = g; }
        else                { accum[idx] += g; }
    }
}

// ---------------------------------------------------------------------------
// Row-normalize hop mean: z = (hsum*invT) / max(||.||, eps). One wave per row
// (COUT==32 == wave32). Writes z (f32, to d_out tail) and zf16 for z@z^T.
// Exact division kept here (only 131K elements, accuracy matters for z).
// ---------------------------------------------------------------------------
__global__ void normalize_rows(const float* __restrict__ hsum, float invT,
                               float* __restrict__ zout, _Float16* __restrict__ zf16)
{
    const int i = blockIdx.x;
    const int c = threadIdx.x;              // 32 cols
    const float v = hsum[(size_t)i * 32 + c] * invT;
    float sq = v * v;
    #pragma unroll
    for (int off = 16; off; off >>= 1) sq += __shfl_xor(sq, off, 32);
    float nrm = sqrtf(sq);
    nrm = nrm > EPSN ? nrm : EPSN;
    const float z = v / nrm;
    zout[(size_t)i * 32 + c] = z;
    zf16[(size_t)i * 32 + c] = (_Float16)z;
}

// ---------------------------------------------------------------------------
// A_pred = sigmoid(z @ z^T). K=32 -> one WMMA per 16x16 tile. B-operand rows
// of z are contiguous in K (z^T[k,j] == z[j,k]). Sigmoid via v_exp_f32 +
// v_rcp_f32 (both TRANS ops). Output is write-once -> nontemporal stores
// keep M_t resident in the 192MB L2 between layers.
// ---------------------------------------------------------------------------
__global__ void zzt_sigmoid(const _Float16* __restrict__ zf16, float* __restrict__ A)
{
    const int lane = threadIdx.x & 31;
    const int wave = threadIdx.x >> 5;
    const int i0   = blockIdx.x * 16;
    const int j0   = blockIdx.y * 64 + wave * 16;
    const int m    = lane & 15;
    const int half = lane >> 4;

    v16h a;
    const _Float16* ar = &zf16[(size_t)(i0 + m) * 32 + half * 16];
    #pragma unroll
    for (int t = 0; t < 16; ++t) a[t] = ar[t];
    const v16h b = *(const v16h*)&zf16[(size_t)(j0 + m) * 32 + half * 16];
    v8f acc = {};
    acc = wmma_f16(a, b, acc);

    #pragma unroll
    for (int r = 0; r < 8; ++r) {
        const int row = i0 + r + (half ? 8 : 0);
        const int col = j0 + m;
        const float sig = fastrcp(1.f + __expf(-acc[r]));
        __builtin_nontemporal_store(sig, &A[(size_t)row * NROWS + col]);
    }
}

// ---------------------------------------------------------------------------
extern "C" void kernel_launch(void* const* d_in, const int* in_sizes, int n_in,
                              void* d_out, int out_size, void* d_ws, size_t ws_size,
                              hipStream_t stream)
{
    constexpr int N = NROWS, CIN = 128, CH = 64, COUT = 32;

    const float* x    = (const float*)d_in[0];
    const float* adj  = (const float*)d_in[1];
    const float* Mh   = (const float*)d_in[2];   // [T,N,N]
    const float* W1   = (const float*)d_in[3];
    const float* a_s1 = (const float*)d_in[4];
    const float* a_n1 = (const float*)d_in[5];
    const float* W2   = (const float*)d_in[6];
    const float* a_s2 = (const float*)d_in[7];
    const float* a_n2 = (const float*)d_in[8];

    const int T = in_sizes[2] / (N * N);         // == 3

    float* A_pred = (float*)d_out;               // [N,N]
    float* z_out  = A_pred + (size_t)N * N;      // [N,COUT]

    // workspace carve (256B aligned)
    size_t off = 0;
    auto carve = [&](size_t bytes) -> void* {
        void* p = (char*)d_ws + off;
        off = (off + bytes + 255) & ~(size_t)255;
        return p;
    };
    float*     h1f32 = (float*)    carve((size_t)N * CH   * 4);
    _Float16*  h1f16 = (_Float16*) carve((size_t)N * CH   * 2);
    float*     s1    = (float*)    carve((size_t)N * 4);
    float*     n1    = (float*)    carve((size_t)N * 4);
    float*     Gf32  = (float*)    carve((size_t)N * CH   * 4);
    _Float16*  Gf16  = (_Float16*) carve((size_t)N * CH   * 2);
    float*     h2f32 = (float*)    carve((size_t)N * COUT * 4);
    _Float16*  h2f16 = (_Float16*) carve((size_t)N * COUT * 2);
    float*     s2    = (float*)    carve((size_t)N * 4);
    float*     n2    = (float*)    carve((size_t)N * 4);
    float*     hsum  = (float*)    carve((size_t)N * COUT * 4);
    _Float16*  zf16  = (_Float16*) carve((size_t)N * COUT * 2);
    (void)ws_size; (void)n_in; (void)out_size;

    // h1 = x @ W1 (hop-invariant), then score dots
    gemm_wmma<CIN, CH><<<N / 16, 32 * (CH / 16), 0, stream>>>(x, W1, h1f32, h1f16);
    row_dots<CH><<<N / 256, 256, 0, stream>>>(h1f32, a_s1, a_n1, s1, n1);

    for (int t = 0; t < T; ++t) {
        const float* Mt = Mh + (size_t)t * N * N;
        // layer 1: G_t = elu(softmax(e1_t) @ h1)
        attn_wmma<CH><<<N / 16, 32 * (CH / 16), 0, stream>>>(
            Mt, adj, s1, n1, h1f16, Gf32, Gf16, nullptr, 0);
        // layer 2 features: h2_t = G_t @ W2, then score dots
        gemm_wmma<CH, COUT><<<N / 16, 32 * (COUT / 16), 0, stream>>>(Gf32, W2, h2f32, h2f16);
        row_dots<COUT><<<N / 256, 256, 0, stream>>>(h2f32, a_s2, a_n2, s2, n2);
        // layer 2 attention -> accumulate hop sum
        attn_wmma<COUT><<<N / 16, 32 * (COUT / 16), 0, stream>>>(
            Mt, adj, s2, n2, h2f16, nullptr, nullptr, hsum, t == 0 ? 1 : 2);
    }

    // z = normalize(mean over hops); write z to d_out tail + f16 copy
    normalize_rows<<<N, 32, 0, stream>>>(hsum, 1.0f / (float)T, z_out, zf16);

    // A_pred = sigmoid(z @ z^T)
    zzt_sigmoid<<<dim3(N / 16, N / 64), 128, 0, stream>>>(zf16, A_pred);
}
